// Bond2BondLayer_54889682043681
// MI455X (gfx1250) — matrix operand order repaired and checked by reading, not compiled, not run on hardware
//
#include <hip/hip_runtime.h>

#define NNODES 100000
#define NEDGES 500000
#define NHEADS 4
#define DIM    128           // bond dim == hidden dim
#define OUTSTRIDE (NHEADS * DIM)   // 512
#define NTILES (NEDGES / 16)       // 31250 edge tiles per head
#define WAVES_PER_BLOCK 8
#define BLOCKS_PER_HEAD ((NTILES + WAVES_PER_BLOCK - 1) / WAVES_PER_BLOCK) // 3907

typedef __bf16 v16bf __attribute__((ext_vector_type(16)));
typedef float  v8f   __attribute__((ext_vector_type(8)));

union Frag32B { int4 p[2]; v16bf f; };

// ---- round-to-nearest-even f32 -> bf16 (bit pattern) ----
static __device__ inline unsigned int f2bf(float x) {
    union { float f; unsigned int u; } v; v.f = x;
    unsigned int r = v.u + 0x7fffu + ((v.u >> 16) & 1u);
    return r >> 16;
}

// ---- hardware tanh (CDNA5 V_TANH_F32 transcendental) ----
static __device__ inline float fast_tanh(float x) {
#if __has_builtin(__builtin_amdgcn_tanhf)
    return __builtin_amdgcn_tanhf(x);
#else
    float y;
    asm volatile("v_tanh_f32 %0, %1\n\tv_nop" : "=v"(y) : "v"(x));
    return y;
#endif
}

// ============================================================
// Pack W [4][256][128] f32 -> bf16 WMMA B-fragments.
// frag id = a*64 + kt*8 + nt   (kt: K-block of 32, nt: N-block of 16)
// within frag (256 dwords): dword index = lane*8 + d
//   lane l -> n = nt*16 + (l&15), khalf = (l>>4)*16
//   dword d holds pair (k, k+1) with k = kt*32 + khalf + 2*d
// ============================================================
__global__ void pack_w_kernel(const float* __restrict__ W, unsigned int* __restrict__ wf) {
    int id = blockIdx.x * blockDim.x + threadIdx.x;       // 65536 dwords total
    if (id >= NHEADS * 64 * 256) return;
    int d    = id & 7;
    int l    = (id >> 3) & 31;
    int frag = id >> 8;
    int nt   = frag & 7;
    int kt   = (frag >> 3) & 7;
    int a    = frag >> 6;
    int n = nt * 16 + (l & 15);
    int k = kt * 32 + ((l >> 4) * 16) + 2 * d;
    const float* Wa = W + (size_t)a * 256 * 128;
    unsigned int lo = f2bf(Wa[(size_t)k * 128 + n]);
    unsigned int hi = f2bf(Wa[(size_t)(k + 1) * 128 + n]);
    wf[id] = lo | (hi << 16);
}

// ============================================================
// bond_feat f32 [N,128] -> bf16 row-major (same layout)
// ============================================================
__global__ void conv_bond_kernel(const float4* __restrict__ in, uint2* __restrict__ out) {
    int i = blockIdx.x * blockDim.x + threadIdx.x;        // one float4 each
    if (i >= NNODES * DIM / 4) return;
    float4 f = in[i];
    uint2 o;
    o.x = f2bf(f.x) | (f2bf(f.y) << 16);
    o.y = f2bf(f.z) | (f2bf(f.w) << 16);
    out[i] = o;
}

// ============================================================
// Zero output (we accumulate with atomics; d_out is poisoned)
// ============================================================
__global__ void zero_kernel(float4* __restrict__ out, int n4) {
    for (int i = blockIdx.x * blockDim.x + threadIdx.x; i < n4; i += gridDim.x * blockDim.x)
        out[i] = make_float4(0.f, 0.f, 0.f, 0.f);
}

// ============================================================
// Main fused kernel: per head / per 16-edge tile
//   A = [x_i | x_j]  (16 x 256 bf16),  B = W_head (256 x 128 bf16, LDS)
//   C = A@B (f32)  ->  alpha = tanh(C + b)  ->  atomicAdd(out[dst], alpha*x_j)
// ============================================================
__global__ void __launch_bounds__(256)
bond2bond_kernel(const float* __restrict__ bond,        // f32 [N,128]
                 const int*   __restrict__ ei,          // [4][2][E]
                 const float* __restrict__ bias,        // [4][128]
                 const unsigned short* __restrict__ bond_bf, // bf16 [N,128]
                 const int4*  __restrict__ wfrag,       // packed W fragments
                 float*       __restrict__ out)         // [N,512]
{
    __shared__ int4 sW[4096];                           // 64 KB: one head's W fragments

    const int a = blockIdx.y;

    // ---- Stage this head's packed W into LDS.
    // Prefer the CDNA5 async global->LDS path (bypasses VGPRs, ASYNCcnt-tracked).
    {
        const int4* gW = wfrag + (size_t)a * 4096;
#if __has_builtin(__builtin_amdgcn_global_load_async_to_lds_b128) && \
    __has_builtin(__builtin_amdgcn_s_wait_asynccnt)
        typedef int nat_int4 __attribute__((vector_size(16)));
        typedef __attribute__((address_space(1))) nat_int4 g_nat_int4;
        typedef __attribute__((address_space(3))) nat_int4 l_nat_int4;
        for (int i = threadIdx.x; i < 4096; i += 256) {
            __builtin_amdgcn_global_load_async_to_lds_b128(
                (g_nat_int4*)(gW + i),
                (l_nat_int4*)(&sW[i]),
                0, 0);
        }
        __builtin_amdgcn_s_wait_asynccnt(0);
#else
        for (int i = threadIdx.x; i < 4096; i += 256) sW[i] = gW[i];
#endif
    }
    __syncthreads();

    const int wave = threadIdx.x >> 5;
    const int lane = threadIdx.x & 31;
    const int tile = blockIdx.x * WAVES_PER_BLOCK + wave;
    if (tile >= NTILES) return;

    const int e0   = tile * 16;
    const int m    = lane & 15;      // edge row this lane gathers for A
    const int half = lane >> 4;      // half-wave id (K split: k = 16*half + 0..15 in each 32-step)
    const int e    = e0 + m;

    const int* src_arr = ei + (size_t)a * 2 * NEDGES;            // edge_index[a][0]
    const int* dst_arr = ei + (size_t)a * 2 * NEDGES + NEDGES;   // edge_index[a][1]
    const int srcI = src_arr[e];
    const int dstI = dst_arr[e];

    // ---- Gather A fragments: 8 K-steps of 32 (K 0..127 = x_i, 128..255 = x_j)
    // lane l: chunk0 = K[s*32 + half*16 .. +7], chunk1 = K[s*32 + half*16 + 8 .. +7]
    const unsigned short* xi = bond_bf + (size_t)dstI * DIM;
    const unsigned short* xj = bond_bf + (size_t)srcI * DIM;
    const int koff = half * 16;

    Frag32B A[8];
#pragma unroll
    for (int s = 0; s < 8; ++s) {
        const unsigned short* rowp = (s < 4) ? xi : xj;
        const int kk = (s & 3) * 32 + koff;
        A[s].p[0] = *(const int4*)(rowp + kk);
        A[s].p[1] = *(const int4*)(rowp + kk + 8);
    }

    const int ncol = lane & 15;      // column within an N-tile (C layout: N = lane&15)

    // ---- Hoisted epilogue state: per C-VGPR row v this lane holds row (v + half*8).
    // Redistribute src/dst via wave shuffle ONCE; keep 32-bit element offsets so the
    // backend can use SGPR-base + 32-bit VGPR offset (GVS) addressing.
    unsigned xoff[8];                // element index into bond:  src_row*128 + ncol
    unsigned ooff[8];                // element index into out:   dst_row*512 + a*128 + ncol
#pragma unroll
    for (int v = 0; v < 8; ++v) {
        const int row = v + half * 8;
        const int sm  = __shfl(srcI, row, 32);
        const int dm  = __shfl(dstI, row, 32);
        xoff[v] = (unsigned)sm * DIM + ncol;
        ooff[v] = (unsigned)dm * OUTSTRIDE + (unsigned)(a * DIM + ncol);
    }
    const float* biasp = bias + a * DIM + ncol;

    // ---- 8 N-tiles of 16 columns
#pragma unroll 1
    for (int nt = 0; nt < 8; ++nt) {
        v8f c = {};
#pragma unroll
        for (int s = 0; s < 8; ++s) {
            Frag32B B;
            const int4* bp = &sW[(s * 8 + nt) * 64 + lane * 2];
            B.p[0] = bp[0];
            B.p[1] = bp[1];
            c = __builtin_amdgcn_wmma_f32_16x16x32_bf16(
                    false, A[s].f, false, B.f, (short)0, c, false, false);
        }

        const unsigned off = nt * 16;        // column offset of this N-tile
        const float bsv = biasp[off];        // tiny L0-resident load

#pragma unroll
        for (int v = 0; v < 8; ++v) {
            const float alpha = fast_tanh(c[v] + bsv);
            const float xjv   = bond[xoff[v] + off];          // f32 x_j (coalesced)
            unsafeAtomicAdd(&out[ooff[v] + off], alpha * xjv);
        }
    }
}

extern "C" void kernel_launch(void* const* d_in, const int* in_sizes, int n_in,
                              void* d_out, int out_size, void* d_ws, size_t ws_size,
                              hipStream_t stream) {
    const float* bond = (const float*)d_in[0];   // [100000,128] f32
    const int*   ei   = (const int*)d_in[1];     // [4][2][500000] int
    const float* W    = (const float*)d_in[2];   // [4][256][128] f32
    const float* bias = (const float*)d_in[3];   // [4][128] f32
    float*       out  = (float*)d_out;           // [100000,512] f32

    unsigned char* ws = (unsigned char*)d_ws;
    unsigned int*  wfrag_u  = (unsigned int*)ws;                     // 256 KB packed W
    int4*          wfrag    = (int4*)ws;
    unsigned short* bond_bf = (unsigned short*)(ws + 262144);        // 25.6 MB bf16 feats

    // 1) pack W into bf16 B-fragments (65536 dwords)
    pack_w_kernel<<<(NHEADS * 64 * 256 + 255) / 256, 256, 0, stream>>>(W, wfrag_u);

    // 2) convert bond_feat to bf16 (3.2M float4s)
    conv_bond_kernel<<<(NNODES * DIM / 4 + 255) / 256, 256, 0, stream>>>(
        (const float4*)bond, (uint2*)bond_bf);

    // 3) zero the output accumulator
    zero_kernel<<<2048, 256, 0, stream>>>((float4*)out, out_size / 4);

    // 4) fused gather + WMMA GEMM + tanh + scatter-add
    dim3 grid(BLOCKS_PER_HEAD, NHEADS, 1);
    bond2bond_kernel<<<grid, 256, 0, stream>>>(bond, ei, bias, bond_bf, wfrag, out);
}
